// MixtureOfExperts_996432413517
// MI455X (gfx1250) — matrix operand order repaired and checked by reading, not compiled, not run on hardware
//
#include <hip/hip_runtime.h>
#include <hip/hip_bf16.h>
#include <math.h>

// ---------------------------------------------------------------------------
// MoE forward for MI455X (gfx1250, wave32, WMMA).
//   T=8192 tokens, H=1024, I=4096, E=8, top-2.
// router -> per-expert compacted token lists -> fused SwiGLU FFN per
// (expert, 16-token tile) with v_wmma_f32_16x16x32_f16 (f32 accumulate).
// Weights are pre-converted to f16 in workspace once per launch (if ws is
// large enough), removing per-tile v_cvt traffic and halving weight reads.
// Hidden activations [16 x 4096] f16 live entirely in LDS (~128KB of the
// 320KB WGP LDS).
// ---------------------------------------------------------------------------

#define T_TOKENS 8192
#define H_DIM    1024
#define I_DIM    4096
#define NE       8

#define XS_STRIDE  1032   // 16 rows x (1024+8) halves = 33,024 B
#define HID_STRIDE 4104   // 16 rows x (4096+8) halves = 131,328 B

typedef __attribute__((ext_vector_type(16))) _Float16 v16h;
typedef __attribute__((ext_vector_type(8)))  _Float16 v8h;
typedef __attribute__((ext_vector_type(4)))  _Float16 v4h;
typedef __attribute__((ext_vector_type(8)))  float    v8f;

// ---- WMMA helpers ---------------------------------------------------------

__device__ __forceinline__ v8f wmma_f16(v16h a, v16h b, v8f c) {
  // (neg_a, A, neg_b, B, c_mod, C, reuse_a, reuse_b)
  return __builtin_amdgcn_wmma_f32_16x16x32_f16(false, a, false, b,
                                                (short)0, c, false, false);
}

// A-matrix 16x32 f16: lanes 0-15 hold row M=lane, K=0..15;
// lanes 16-31 hold row M=lane-16, K=16..31 (matches B's contiguous K order).
__device__ __forceinline__ v16h load_a_tile(const _Float16* base, int rstride,
                                            int kb, int lane) {
  const _Float16* p = base + (lane & 15) * rstride + kb + ((lane >> 4) << 4);
  v8h lo = *(const v8h*)(p);
  v8h hi = *(const v8h*)(p + 8);
  v16h a;
#pragma unroll
  for (int j = 0; j < 8; ++j) { a[j] = lo[j]; a[8 + j] = hi[j]; }
  return a;
}

// B-matrix 32x16 f16 (KxN): lane holds column N=lane&15; lanes 0-15 K=0..15,
// lanes 16-31 K=16..31.
// f16 weights: 16 contiguous halves (32B) per lane, straight load.
__device__ __forceinline__ v16h load_b_tile(const _Float16* __restrict__ w,
                                            int ld, int colbase, int kb,
                                            int lane) {
  const v8h* p = (const v8h*)(w + (size_t)(colbase + (lane & 15)) * ld
                                + kb + ((lane >> 4) << 4));
  v8h lo = p[0], hi = p[1];
  v16h b;
#pragma unroll
  for (int j = 0; j < 8; ++j) { b[j] = lo[j]; b[8 + j] = hi[j]; }
  return b;
}

// f32 weights fallback: 16 contiguous floats (64B) per lane, convert in regs.
__device__ __forceinline__ v16h load_b_tile(const float* __restrict__ w,
                                            int ld, int colbase, int kb,
                                            int lane) {
  const float4* p = (const float4*)(w + (size_t)(colbase + (lane & 15)) * ld
                                      + kb + ((lane >> 4) << 4));
  v16h b;
#pragma unroll
  for (int q = 0; q < 4; ++q) {
    float4 v = p[q];
    b[q * 4 + 0] = (_Float16)v.x;
    b[q * 4 + 1] = (_Float16)v.y;
    b[q * 4 + 2] = (_Float16)v.z;
    b[q * 4 + 3] = (_Float16)v.w;
  }
  return b;
}

// ---- Kernel 0: f32 -> f16 weight conversion -------------------------------

__global__ __launch_bounds__(256) void cvt_f16_kernel(
    const float* __restrict__ src, _Float16* __restrict__ dst, size_t n) {
  size_t i = ((size_t)blockIdx.x * blockDim.x + threadIdx.x) * 4;
  const size_t stride = (size_t)gridDim.x * blockDim.x * 4;
  for (; i < n; i += stride) {
    float4 v = *(const float4*)(src + i);
    v4h o = {(_Float16)v.x, (_Float16)v.y, (_Float16)v.z, (_Float16)v.w};
    *(v4h*)(dst + i) = o;
  }
}

// ---- Kernel 1: router -----------------------------------------------------
// One wave (32 lanes) per token: logits -> softmax -> top-2 -> renorm.

__global__ __launch_bounds__(256) void moe_router_kernel(
    const float* __restrict__ x, const float* __restrict__ rw,
    float* __restrict__ usage, int* __restrict__ counts,
    int* __restrict__ toklist, float* __restrict__ wlist) {
  __shared__ float us[NE];
  const int tid = threadIdx.x;
  if (tid < NE) us[tid] = 0.0f;
  __syncthreads();

  const int lane = tid & 31;            // gfx1250 is wave32-only
  const int wave = tid >> 5;
  const int t = blockIdx.x * 8 + wave;

  float xv[32];
  const float* xp = x + (size_t)t * H_DIM + lane;
#pragma unroll
  for (int j = 0; j < 32; ++j) xv[j] = xp[j * 32];

  float logits[NE];
#pragma unroll
  for (int e = 0; e < NE; ++e) {
    const float* wp = rw + e * H_DIM + lane;
    float acc = 0.0f;
#pragma unroll
    for (int j = 0; j < 32; ++j) acc += xv[j] * wp[j * 32];
#pragma unroll
    for (int off = 16; off > 0; off >>= 1) acc += __shfl_xor(acc, off, 32);
    logits[e] = acc;
  }

  if (lane == 0) {
    float mx = logits[0];
#pragma unroll
    for (int e = 1; e < NE; ++e) mx = fmaxf(mx, logits[e]);
    float p[NE], s = 0.0f;
#pragma unroll
    for (int e = 0; e < NE; ++e) { p[e] = __expf(logits[e] - mx); s += p[e]; }
    float inv = 1.0f / s;
#pragma unroll
    for (int e = 0; e < NE; ++e) {
      p[e] *= inv;
      atomicAdd(&us[e], p[e]);
    }
    int i1 = 0;                          // ties -> lowest index (lax.top_k)
#pragma unroll
    for (int e = 1; e < NE; ++e) if (p[e] > p[i1]) i1 = e;
    int i2 = (i1 == 0) ? 1 : 0;
#pragma unroll
    for (int e = 0; e < NE; ++e) if (e != i1 && p[e] > p[i2]) i2 = e;
    float sw = 1.0f / (p[i1] + p[i2]);
    int pos1 = atomicAdd(&counts[i1], 1);
    toklist[i1 * T_TOKENS + pos1] = t;
    wlist[i1 * T_TOKENS + pos1] = p[i1] * sw;
    int pos2 = atomicAdd(&counts[i2], 1);
    toklist[i2 * T_TOKENS + pos2] = t;
    wlist[i2 * T_TOKENS + pos2] = p[i2] * sw;
  }
  __syncthreads();
  if (tid < NE) atomicAdd(&usage[tid], us[tid]);
}

// ---- Kernel 2: aux load-balancing loss ------------------------------------

__global__ void moe_aux_kernel(const float* __restrict__ usage,
                               float* __restrict__ out_aux) {
  if (threadIdx.x == 0) {
    float u[NE], s = 0.0f;
#pragma unroll
    for (int e = 0; e < NE; ++e) { u[e] = usage[e]; s += u[e]; }
    float inv = 1.0f / s, loss = 0.0f;
#pragma unroll
    for (int e = 0; e < NE; ++e) {
      float d = u[e] * inv - (1.0f / NE);
      loss += d * d;
    }
    out_aux[0] = 0.01f * (loss / NE);
  }
}

// ---- Kernel 3: fused per-expert SwiGLU FFN --------------------------------
// Grid: (T/16, E). Block 256 = 8 waves; each block owns 16 tokens of one
// expert's list. Per wave iteration: one A-tile (from LDS) feeds 4 WMMAs
// (2 N-tiles x gate/up) in stage 1 and 2 WMMAs in stage 2 — 4 (resp. 2)
// independent accumulator chains keep the WMMA pipe free of RAW stalls.

template <typename WT>
__global__ __launch_bounds__(256) void moe_ffn_kernel(
    const float* __restrict__ x,
    const WT* __restrict__ wg, const WT* __restrict__ wu,
    const WT* __restrict__ wd,
    const int* __restrict__ counts, const int* __restrict__ toklist,
    const float* __restrict__ wlist, float* __restrict__ out) {
  const int e = blockIdx.y;
  const int nt = counts[e];
  const int row0 = blockIdx.x * 16;
  if (row0 >= nt) return;   // uniform per block: safe before barriers

  extern __shared__ char smem_raw[];
  _Float16* Xs  = (_Float16*)smem_raw;                 // 16 x XS_STRIDE
  _Float16* Hid = Xs + 16 * XS_STRIDE;                 // 16 x HID_STRIDE
  int*      tokS = (int*)(Hid + 16 * HID_STRIDE);      // 16
  float*    cwS  = (float*)(tokS + 16);                // 16

  const int tid = threadIdx.x;
  const int lane = tid & 31;
  const int wave = tid >> 5;

  if (tid < 16) {
    int r = row0 + tid;
    tokS[tid] = (r < nt) ? toklist[e * T_TOKENS + r] : 0;
    cwS[tid]  = (r < nt) ? wlist[e * T_TOKENS + r] : 0.0f;
  }
  __syncthreads();

  // Gather 16 token rows of x into LDS as f16.
  for (int idx = tid; idx < 16 * H_DIM; idx += 256) {
    int m = idx >> 10;            // H_DIM = 1024
    int h = idx & (H_DIM - 1);
    Xs[m * XS_STRIDE + h] = (_Float16)x[(size_t)tokS[m] * H_DIM + h];
  }
  __syncthreads();

  const WT* wg_e = wg + (size_t)e * I_DIM * H_DIM;
  const WT* wu_e = wu + (size_t)e * I_DIM * H_DIM;
  const int n  = lane & 15;
  const int mh = (lane >> 4) << 3;   // C/D layout: VGPR r -> M = r + 8*(lane>=16)

  // ---- Stage 1: gate/up GEMMs + SwiGLU -> Hid (f16 in LDS) ----
  for (int ct = wave; ct < I_DIM / 32; ct += 8) {
    const int ic0 = ct << 5;
    const int ic1 = ic0 + 16;
    v8f cg0 = {}, cu0 = {}, cg1 = {}, cu1 = {};
    for (int kc = 0; kc < H_DIM; kc += 32) {
      if (kc + 32 < H_DIM) {  // prefetch next K-chunk of weights
        __builtin_prefetch(wg_e + (size_t)(ic0 + n) * H_DIM + kc + 32, 0, 3);
        __builtin_prefetch(wu_e + (size_t)(ic0 + n) * H_DIM + kc + 32, 0, 3);
      }
      v16h a   = load_a_tile(Xs, XS_STRIDE, kc, lane);
      v16h bg0 = load_b_tile(wg_e, H_DIM, ic0, kc, lane);
      v16h bu0 = load_b_tile(wu_e, H_DIM, ic0, kc, lane);
      v16h bg1 = load_b_tile(wg_e, H_DIM, ic1, kc, lane);
      v16h bu1 = load_b_tile(wu_e, H_DIM, ic1, kc, lane);
      cg0 = wmma_f16(a, bg0, cg0);
      cu0 = wmma_f16(a, bu0, cu0);
      cg1 = wmma_f16(a, bg1, cg1);
      cu1 = wmma_f16(a, bu1, cu1);
    }
#pragma unroll
    for (int r = 0; r < 8; ++r) {
      float g0 = cg0[r], u0 = cu0[r];
      float g1 = cg1[r], u1 = cu1[r];
      Hid[(mh + r) * HID_STRIDE + ic0 + n] =
          (_Float16)((g0 / (1.0f + __expf(-g0))) * u0);
      Hid[(mh + r) * HID_STRIDE + ic1 + n] =
          (_Float16)((g1 / (1.0f + __expf(-g1))) * u1);
    }
  }
  __syncthreads();

  // ---- Stage 2: down projection + weighted scatter-add combine ----
  const WT* wd_e = wd + (size_t)e * H_DIM * I_DIM;
  for (int ct = wave; ct < H_DIM / 32; ct += 8) {
    const int hc0 = ct << 5;
    const int hc1 = hc0 + 16;
    v8f cy0 = {}, cy1 = {};
    for (int kc = 0; kc < I_DIM; kc += 32) {
      if (kc + 32 < I_DIM)
        __builtin_prefetch(wd_e + (size_t)(hc0 + n) * I_DIM + kc + 32, 0, 3);
      v16h a  = load_a_tile(Hid, HID_STRIDE, kc, lane);
      v16h b0 = load_b_tile(wd_e, I_DIM, hc0, kc, lane);
      v16h b1 = load_b_tile(wd_e, I_DIM, hc1, kc, lane);
      cy0 = wmma_f16(a, b0, cy0);
      cy1 = wmma_f16(a, b1, cy1);
    }
#pragma unroll
    for (int r = 0; r < 8; ++r) {
      int m = mh + r;
      float w = cwS[m];
      if (w != 0.0f) {
        atomicAdd(&out[(size_t)tokS[m] * H_DIM + hc0 + n], w * cy0[r]);
        atomicAdd(&out[(size_t)tokS[m] * H_DIM + hc1 + n], w * cy1[r]);
      }
    }
  }
}

// ---- launch ---------------------------------------------------------------

extern "C" void kernel_launch(void* const* d_in, const int* in_sizes, int n_in,
                              void* d_out, int out_size, void* d_ws, size_t ws_size,
                              hipStream_t stream) {
  const float* x  = (const float*)d_in[0];   // [T, H]
  const float* rw = (const float*)d_in[1];   // [E, H]
  const float* wg = (const float*)d_in[2];   // [E, I, H]
  const float* wu = (const float*)d_in[3];   // [E, I, H]
  const float* wd = (const float*)d_in[4];   // [E, H, I]
  float* out = (float*)d_out;                // [T*H] + [1] aux

  // Workspace layout:
  //  [0,32)      usage: 8 f32
  //  [32,64)     counts: 8 i32
  //  [64,..)     toklist: E*T i32 (256KB), wlist: E*T f32 (256KB)
  //  [aligned..) optional f16 weight copies: 3 x E*I*H halves (64MB each)
  float* usage   = (float*)d_ws;
  int*   counts  = (int*)((char*)d_ws + 32);
  int*   toklist = (int*)((char*)d_ws + 64);
  float* wlist   = (float*)((char*)d_ws + 64 + (size_t)NE * T_TOKENS * sizeof(int));

  const size_t W_ELEMS  = (size_t)NE * I_DIM * H_DIM;      // 33.5M per tensor
  const size_t lists_end = 64 + (size_t)NE * T_TOKENS * 8;
  const size_t w16_off   = (lists_end + 255) & ~(size_t)255;
  const size_t need16    = w16_off + 3 * W_ELEMS * sizeof(_Float16);

  hipMemsetAsync(d_ws, 0, 64, stream);                                  // usage+counts
  hipMemsetAsync(d_out, 0, (size_t)T_TOKENS * H_DIM * sizeof(float), stream);

  moe_router_kernel<<<T_TOKENS / 8, 256, 0, stream>>>(x, rw, usage, counts,
                                                      toklist, wlist);
  moe_aux_kernel<<<1, 32, 0, stream>>>(usage, out + (size_t)T_TOKENS * H_DIM);

  const size_t smem = (size_t)(16 * XS_STRIDE + 16 * HID_STRIDE) * sizeof(_Float16)
                    + 16 * sizeof(int) + 16 * sizeof(float);   // ~164.5 KB WGP LDS
  dim3 grid(T_TOKENS / 16, NE);

  if (ws_size >= need16) {
    // Pre-convert weights to f16 once per launch (~600MB of HBM traffic,
    // ~26us at 23.3 TB/s; removes all v_cvt from the WMMA hot loops and
    // halves the streamed weight bytes).
    _Float16* wg16 = (_Float16*)((char*)d_ws + w16_off);
    _Float16* wu16 = wg16 + W_ELEMS;
    _Float16* wd16 = wu16 + W_ELEMS;
    cvt_f16_kernel<<<8192, 256, 0, stream>>>(wg, wg16, W_ELEMS);
    cvt_f16_kernel<<<8192, 256, 0, stream>>>(wu, wu16, W_ELEMS);
    cvt_f16_kernel<<<8192, 256, 0, stream>>>(wd, wd16, W_ELEMS);
    moe_ffn_kernel<_Float16><<<grid, 256, smem, stream>>>(
        x, wg16, wu16, wd16, counts, toklist, wlist, out);
  } else {
    // Fallback: convert f32 weights in registers while streaming.
    moe_ffn_kernel<float><<<grid, 256, smem, stream>>>(
        x, wg, wu, wd, counts, toklist, wlist, out);
  }
}
